// SimpleSLSTM_27977416966481
// MI455X (gfx1250) — matrix operand order repaired and checked by reading, not compiled, not run on hardware
//
#include <hip/hip_runtime.h>
#include <hip/hip_bf16.h>
#include <stddef.h>
#include <stdint.h>

// ---------------------------------------------------------------------------
// Model dims (from reference)
// ---------------------------------------------------------------------------
#define BATCH   4096
#define KNEI    8
#define OBS     10
#define PRED    25
#define INDIM   7
#define HDIM    256
#define OUTDIM  2
#define GATES   1024          // 4*H
#define KTILES  9             // augmented K = 288 = 9 * 32
#define APAD    296           // padded LDS A row length in halfs (bank-conflict free)
#define NTILES  64            // 1024 / 16 column tiles
#define MROWS   32            // batch rows per workgroup
#define THREADS 512           // 16 waves
#define FRAG_HALFS (KTILES * NTILES * 32 * 16)   // per-LSTM swizzled weight fragments

typedef _Float16 v16h __attribute__((ext_vector_type(16)));
typedef _Float16 v8h  __attribute__((ext_vector_type(8)));
typedef float    v8f  __attribute__((ext_vector_type(8)));

union AV { v16h v; v8h h[2]; };

static __device__ __forceinline__ v8f vzero8() {
    v8f z;
#pragma unroll
    for (int i = 0; i < 8; ++i) z[i] = 0.f;
    return z;
}
static __device__ __forceinline__ float sigf(float x) {
    return 1.f / (1.f + __expf(-x));
}
// Opaque zero offset, re-materialized every loop iteration.  Adding it to the
// fragment pointer makes the weight loads loop-variant (defeats LICM, which
// otherwise hoists 576 VGPRs worth of loads and spills to scratch) while
// PRESERVING the pointer's global address-space provenance, so the loads
// lower to global_load_b128 (LOADcnt only) instead of flat_load (LOADcnt +
// DScnt, which serializes against the LDS A-matrix stream).
static __device__ __forceinline__ size_t opaque_zero() {
    size_t z = 0;
    asm volatile("" : "+s"(z));
    return z;
}

// Load the four gate fragments of one k-tile into distinct registers.
static __device__ __forceinline__ void load_b4(AV b[4], const _Float16* fragp,
                                               int kt, int wave, int lane) {
#pragma unroll
    for (int g = 0; g < 4; ++g) {
        const _Float16* bf =
            fragp + (((size_t)(kt * NTILES + g * 16 + wave)) * 32 + lane) * 16;
        b[g].h[0] = *(const v8h*)(bf);
        b[g].h[1] = *(const v8h*)(bf + 8);
    }
}

// ---------------------------------------------------------------------------
// Pre-swizzle LSTM weights into per-lane WMMA B-fragment layout (f16).
// Augmented B[k][col]: k<256 -> Whh[col][k]; k==256+d (d<inDim) -> Wih[col][d];
//                      k==256+inDim -> bias[col]; else 0.
// Fragment (kt, nt): lane l holds n = l&15, h = l>>4; 16 k (i<8: 8h+i, else 16+8h+i-8)
// halfs per lane -> 32 contiguous bytes per lane.
// ---------------------------------------------------------------------------
__global__ void prep_frags(const float* __restrict__ Whh,   // [1024][256]
                           const float* __restrict__ Wih,   // [1024][inDim]
                           const float* __restrict__ bias,  // [1024]
                           int inDim, _Float16* __restrict__ frag) {
    int kt = blockIdx.x >> 6;          // 0..8
    int nt = blockIdx.x & 63;          // 0..63
    int lane = threadIdx.x;            // 0..31
    int n = lane & 15, hi = lane >> 4;
    int col = nt * 16 + n;
    int kbase = kt * 32 + hi * 16;
    _Float16* dst = frag + (((size_t)(kt * NTILES + nt)) * 32 + lane) * 16;
#pragma unroll
    for (int i = 0; i < 16; ++i) {
        int k = kt * 32 + (i < 8 ? hi * 8 + i : 16 + hi * 8 + (i - 8));
        float v;
        if (k < HDIM) {
            v = Whh[(size_t)col * HDIM + k];
        } else {
            int d = k - HDIM;
            if (d < inDim)        v = Wih[(size_t)col * inDim + d];
            else if (d == inDim)  v = bias[col];
            else                  v = 0.f;
        }
        dst[i] = (_Float16)v;
    }
}

// ---------------------------------------------------------------------------
// Recurrent LSTM encoder over x:(N,T,inDim); 32 batch rows per workgroup,
// 512 threads (16 waves).  Wave w owns hidden tile w (cols [16w,16w+16)) and
// both row tiles, so each B fragment feeds two WMMAs.
// pool==0: write final h (N,256).  pool==1: mean over groups of 8 rows.
// ---------------------------------------------------------------------------
__global__ void __launch_bounds__(THREADS, 1)
lstm_encode(const float* __restrict__ x, int T, int inDim,
            const _Float16* __restrict__ frag,
            float* __restrict__ hOut, int pool) {
    __shared__ __align__(16) _Float16 hA[MROWS * APAD];

    const int rowBase = blockIdx.x * MROWS;
    const int tid  = threadIdx.x;
    const int wave = tid >> 5;         // 0..15 == hidden tile
    const int lane = tid & 31;
    const int n  = lane & 15;
    const int hi = lane >> 4;

    for (int i = tid; i < MROWS * APAD; i += THREADS) hA[i] = (_Float16)0.f;
    __syncthreads();
    if (tid < MROWS) hA[tid * APAD + HDIM + inDim] = (_Float16)1.f;  // bias col

    v8f cst[2]  = { vzero8(), vzero8() };   // cell state, per row tile
    v8f hnew[2] = { vzero8(), vzero8() };

    for (int t = 0; t < T; ++t) {
        const _Float16* fragp = frag + opaque_zero();
        // inject x_t into augmented columns 256..256+inDim-1
        for (int i = tid; i < MROWS * inDim; i += THREADS) {
            int m = i / inDim, d = i - m * inDim;
            hA[m * APAD + HDIM + d] =
                (_Float16)x[((size_t)(rowBase + m) * T + t) * inDim + d];
        }
        // kt=0 weight fragments do not depend on the LDS slab: issue the
        // loads BEFORE the barrier so they overlap the barrier wait.
        AV bcur[4];
        load_b4(bcur, fragp, 0, wave, lane);
        __syncthreads();

        v8f acc[2][4];
#pragma unroll
        for (int rt = 0; rt < 2; ++rt)
#pragma unroll
            for (int g = 0; g < 4; ++g) acc[rt][g] = vzero8();

#pragma unroll
        for (int kt = 0; kt < KTILES; ++kt) {
            // double-buffer: next k-tile's fragments in flight during WMMAs
            AV bnext[4];
            if (kt + 1 < KTILES) load_b4(bnext, fragp, kt + 1, wave, lane);
            AV a[2];
#pragma unroll
            for (int rt = 0; rt < 2; ++rt) {
                const _Float16* arow =
                    &hA[(rt * 16 + n) * APAD + kt * 32 + hi * 8];
                a[rt].h[0] = *(const v8h*)(arow);
                a[rt].h[1] = *(const v8h*)(arow + 16);
            }
#pragma unroll
            for (int g = 0; g < 4; ++g)
#pragma unroll
                for (int rt = 0; rt < 2; ++rt)
                    acc[rt][g] = __builtin_amdgcn_wmma_f32_16x16x32_f16(
                        false, a[rt].v, false, bcur[g].v, (short)0, acc[rt][g],
                        false, false);
            if (kt + 1 < KTILES) {
#pragma unroll
                for (int g = 0; g < 4; ++g) bcur[g] = bnext[g];
            }
        }
        __syncthreads();   // all waves done reading hA

        // LSTM cell update (gates i,f,g,o co-located per lane/slot)
#pragma unroll
        for (int rt = 0; rt < 2; ++rt) {
#pragma unroll
            for (int r = 0; r < 8; ++r) {
                float ig = sigf(acc[rt][0][r]);
                float fg = sigf(acc[rt][1][r]);
                float gg = tanhf(acc[rt][2][r]);
                float og = sigf(acc[rt][3][r]);
                float cv = fg * cst[rt][r] + ig * gg;
                cst[rt][r]  = cv;
                hnew[rt][r] = og * tanhf(cv);
            }
            int colBase = wave * 16 + n;
#pragma unroll
            for (int r = 0; r < 8; ++r)
                hA[(rt * 16 + hi * 8 + r) * APAD + colBase] = (_Float16)hnew[rt][r];
        }
        __syncthreads();   // hnew stores visible before next iteration's reads
    }

    if (!pool) {
        int col = wave * 16 + n;
#pragma unroll
        for (int rt = 0; rt < 2; ++rt)
#pragma unroll
            for (int r = 0; r < 8; ++r)
                hOut[(size_t)(rowBase + rt * 16 + hi * 8 + r) * HDIM + col] =
                    hnew[rt][r];
    } else {
        // slab rows [16rt+8hi, +8) form neighbor group blockIdx.x*4 + rt*2 + hi
#pragma unroll
        for (int rt = 0; rt < 2; ++rt) {
            float s = 0.f;
#pragma unroll
            for (int r = 0; r < 8; ++r) s += hnew[rt][r];
            s *= 0.125f;
            int grp = blockIdx.x * 4 + rt * 2 + hi;
            hOut[(size_t)grp * HDIM + wave * 16 + n] = s;
        }
    }
}

// ---------------------------------------------------------------------------
// h0 = tanh(concat(h_t, pooled) @ init_W.T + init_b) : 16 rows / block
// ---------------------------------------------------------------------------
__global__ void __launch_bounds__(256, 1)
init_state(const float* __restrict__ hT, const float* __restrict__ pooled,
           const float* __restrict__ W,    // [256][512]
           const float* __restrict__ bvec, // [256]
           float* __restrict__ h0) {
    __shared__ float xin[16][512];
    int rb = blockIdx.x * 16;
    int tid = threadIdx.x;
    for (int i = tid; i < 16 * HDIM; i += 256) {
        int m = i >> 8, j = i & 255;
        xin[m][j]        = hT[(size_t)(rb + m) * HDIM + j];
        xin[m][256 + j]  = pooled[(size_t)(rb + m) * HDIM + j];
    }
    __syncthreads();
    int j = tid;
    float s[16];
#pragma unroll
    for (int m = 0; m < 16; ++m) s[m] = bvec[j];
    for (int k = 0; k < 512; ++k) {
        float w = W[(size_t)j * 512 + k];
#pragma unroll
        for (int m = 0; m < 16; ++m) s[m] += xin[m][k] * w;
    }
#pragma unroll
    for (int m = 0; m < 16; ++m)
        h0[(size_t)(rb + m) * HDIM + j] = tanhf(s[m]);
}

// ---------------------------------------------------------------------------
// Decoder: 25 steps, input = previous step output (dim 2), emits (B,25,2)
// ---------------------------------------------------------------------------
__global__ void __launch_bounds__(THREADS, 1)
lstm_decode(const float* __restrict__ h0, const _Float16* __restrict__ frag,
            const float* __restrict__ outW,   // [2][256]
            const float* __restrict__ outB,   // [2]
            float* __restrict__ out) {        // [B][25][2]
    __shared__ __align__(16) _Float16 hA[MROWS * APAD];
    __shared__ float oAcc[MROWS * 2];   // 32 rows x 2 outputs

    const int rowBase = blockIdx.x * MROWS;
    const int tid  = threadIdx.x;
    const int wave = tid >> 5;
    const int lane = tid & 31;
    const int n  = lane & 15;
    const int hi = lane >> 4;
    const int inDim = OUTDIM;

    for (int i = tid; i < MROWS * APAD; i += THREADS) hA[i] = (_Float16)0.f;
    if (tid < MROWS * 2) oAcc[tid] = 0.f;
    __syncthreads();
    for (int i = tid; i < MROWS * HDIM; i += THREADS) {
        int m = i >> 8, j = i & 255;
        hA[m * APAD + j] = (_Float16)h0[(size_t)(rowBase + m) * HDIM + j];
    }
    if (tid < MROWS) hA[tid * APAD + HDIM + inDim] = (_Float16)1.f;

    // output-projection weights for this lane's hidden column (j = 16*wave+n)
    float wout[2];
    {
        int j = wave * 16 + n;
        wout[0] = outW[j];
        wout[1] = outW[HDIM + j];
    }

    v8f cst[2] = { vzero8(), vzero8() };

    for (int t = 0; t < PRED; ++t) {
        const _Float16* fragp = frag + opaque_zero();
        AV bcur[4];
        load_b4(bcur, fragp, 0, wave, lane);   // overlap with barrier wait
        __syncthreads();   // hA (h + x cols) and zeroed oAcc visible

        v8f acc[2][4];
#pragma unroll
        for (int rt = 0; rt < 2; ++rt)
#pragma unroll
            for (int g = 0; g < 4; ++g) acc[rt][g] = vzero8();

#pragma unroll
        for (int kt = 0; kt < KTILES; ++kt) {
            AV bnext[4];
            if (kt + 1 < KTILES) load_b4(bnext, fragp, kt + 1, wave, lane);
            AV a[2];
#pragma unroll
            for (int rt = 0; rt < 2; ++rt) {
                const _Float16* arow =
                    &hA[(rt * 16 + n) * APAD + kt * 32 + hi * 8];
                a[rt].h[0] = *(const v8h*)(arow);
                a[rt].h[1] = *(const v8h*)(arow + 16);
            }
#pragma unroll
            for (int g = 0; g < 4; ++g)
#pragma unroll
                for (int rt = 0; rt < 2; ++rt)
                    acc[rt][g] = __builtin_amdgcn_wmma_f32_16x16x32_f16(
                        false, a[rt].v, false, bcur[g].v, (short)0, acc[rt][g],
                        false, false);
            if (kt + 1 < KTILES) {
#pragma unroll
                for (int g = 0; g < 4; ++g) bcur[g] = bnext[g];
            }
        }
        __syncthreads();

        v8f hnew[2];
#pragma unroll
        for (int rt = 0; rt < 2; ++rt) {
#pragma unroll
            for (int r = 0; r < 8; ++r) {
                float ig = sigf(acc[rt][0][r]);
                float fg = sigf(acc[rt][1][r]);
                float gg = tanhf(acc[rt][2][r]);
                float og = sigf(acc[rt][3][r]);
                float cv = fg * cst[rt][r] + ig * gg;
                cst[rt][r]  = cv;
                hnew[rt][r] = og * tanhf(cv);
            }
            int colBase = wave * 16 + n;
#pragma unroll
            for (int r = 0; r < 8; ++r)
                hA[(rt * 16 + hi * 8 + r) * APAD + colBase] = (_Float16)hnew[rt][r];
        }

        // output projection: partial dot over this lane's hidden column,
        // reduce over the 16 lanes of each half, then one atomic per wave.
        float po[32];
#pragma unroll
        for (int rt = 0; rt < 2; ++rt)
#pragma unroll
            for (int r = 0; r < 8; ++r)
#pragma unroll
                for (int o = 0; o < 2; ++o)
                    po[(rt * 8 + r) * 2 + o] = hnew[rt][r] * wout[o];
#pragma unroll
        for (int mask = 1; mask < 16; mask <<= 1)
#pragma unroll
            for (int p = 0; p < 32; ++p)
                po[p] += __shfl_xor(po[p], mask, 32);
        if (n == 0) {
#pragma unroll
            for (int rt = 0; rt < 2; ++rt)
#pragma unroll
                for (int r = 0; r < 8; ++r)
#pragma unroll
                    for (int o = 0; o < 2; ++o)
                        atomicAdd(&oAcc[(rt * 16 + hi * 8 + r) * 2 + o],
                                  po[(rt * 8 + r) * 2 + o]);
        }
        __syncthreads();

        if (tid < MROWS * 2) {
            int m = tid >> 1, o = tid & 1;
            float v = oAcc[tid] + outB[o];
            out[((size_t)(rowBase + m) * PRED + t) * OUTDIM + o] = v;
            hA[m * APAD + HDIM + o] = (_Float16)v;  // next step's input
            oAcc[tid] = 0.f;
        }
    }
}

// ---------------------------------------------------------------------------
// Launcher
// ---------------------------------------------------------------------------
extern "C" void kernel_launch(void* const* d_in, const int* in_sizes, int n_in,
                              void* d_out, int out_size, void* d_ws, size_t ws_size,
                              hipStream_t stream) {
    (void)in_sizes; (void)n_in; (void)out_size; (void)ws_size;

    const float* target    = (const float*)d_in[0];
    const float* neigh_dyn = (const float*)d_in[1];
    // d_in[2] neigh_spatial, d_in[3] lane : unused by reference
    const float* enc_Wih = (const float*)d_in[4];
    const float* enc_Whh = (const float*)d_in[5];
    const float* enc_b   = (const float*)d_in[6];
    const float* nei_Wih = (const float*)d_in[7];
    const float* nei_Whh = (const float*)d_in[8];
    const float* nei_b   = (const float*)d_in[9];
    const float* dec_Wih = (const float*)d_in[10];
    const float* dec_Whh = (const float*)d_in[11];
    const float* dec_b   = (const float*)d_in[12];
    const float* init_W  = (const float*)d_in[13];
    const float* init_b  = (const float*)d_in[14];
    const float* out_W   = (const float*)d_in[15];
    const float* out_b   = (const float*)d_in[16];

    char* ws = (char*)d_ws;
    const size_t FRAG_BYTES = (size_t)FRAG_HALFS * 2;           // 589824
    _Float16* encFrag = (_Float16*)(ws);
    _Float16* neiFrag = (_Float16*)(ws + FRAG_BYTES);
    _Float16* decFrag = (_Float16*)(ws + 2 * FRAG_BYTES);
    float* hT     = (float*)(ws + 3 * FRAG_BYTES);                          // 4096*256
    float* pooled = (float*)(ws + 3 * FRAG_BYTES + (size_t)BATCH * HDIM * 4);
    float* h0     = (float*)(ws + 3 * FRAG_BYTES + (size_t)BATCH * HDIM * 8);

    dim3 prepGrid(KTILES * NTILES);
    prep_frags<<<prepGrid, 32, 0, stream>>>(enc_Whh, enc_Wih, enc_b, INDIM, encFrag);
    prep_frags<<<prepGrid, 32, 0, stream>>>(nei_Whh, nei_Wih, nei_b, INDIM, neiFrag);
    prep_frags<<<prepGrid, 32, 0, stream>>>(dec_Whh, dec_Wih, dec_b, OUTDIM, decFrag);

    lstm_encode<<<BATCH / MROWS, THREADS, 0, stream>>>(target, OBS, INDIM,
                                                       encFrag, hT, 0);
    lstm_encode<<<(BATCH * KNEI) / MROWS, THREADS, 0, stream>>>(neigh_dyn, OBS, INDIM,
                                                                neiFrag, pooled, 1);
    init_state<<<BATCH / 16, 256, 0, stream>>>(hT, pooled, init_W, init_b, h0);
    lstm_decode<<<BATCH / MROWS, THREADS, 0, stream>>>(h0, decFrag, out_W, out_b,
                                                       (float*)d_out);
}
